// Mamba_89730456748194
// MI455X (gfx1250) — compile-verified, matrix operand order checked by reading, not run on hardware
//
#include <hip/hip_runtime.h>
#include <hip/hip_bf16.h>
#include <cstdint>
#include <cstddef>

// ---------------------------------------------------------------------------
// MI455X (gfx1250) implementation of the bidirectional Mamba reference.
//  * All GEMMs run on V_WMMA_F32_16X16X32_F16 (fp16 in, fp32 accumulate).
//  * W tiles are staged into LDS with GLOBAL_LOAD_ASYNC_TO_LDS_B128
//    (double-buffered, s_wait_asynccnt), shared by the 4 waves of a block:
//    4x less global W traffic, async copy overlapped with WMMA.
//  * dA / dBu (b,l,d,n) tensors are never materialized: fused into the scan.
//  * GEMM epilogues fuse bias, ReLU, time-reversed residual accumulation and
//    the (b,d,l)->(b,l,d) transposed scatter of the MLP.
// Model is tiny (~70 GFLOP, all tensors fit in the 192MB L2) => latency bound;
// the win is fusion + tensor-path GEMMs + async staging, not exotic tiling.
// ---------------------------------------------------------------------------

typedef _Float16 v16h __attribute__((ext_vector_type(16)));
typedef float    v8f  __attribute__((ext_vector_type(8)));

#define B_   8
#define L_   1024
#define V_   128
#define DM   256      // d_model
#define DI   512      // d_inner
#define DS_  16       // d_state
#define DTR  16       // dt_rank
#define HID_ 1024
#define NL   4

static __device__ __forceinline__ float wave_sum32(float v) {
#pragma unroll
  for (int o = 16; o > 0; o >>= 1) v += __shfl_xor(v, o, 32);
  return v;
}

static __device__ __forceinline__ float silu_f(float x) {
  return x / (1.0f + __expf(-x));
}
static __device__ __forceinline__ float softplus_f(float x) {
  return (x > 20.0f) ? x : log1pf(__expf(x));
}

// raw LDS byte offset (generic -> addrspace(3) -> int) for async-to-LDS ops
static __device__ __forceinline__ uint32_t lds_addr_of(const void* p) {
  return (uint32_t)(uintptr_t)(__attribute__((address_space(3))) const void*)p;
}

// ---------------------------------------------------------------------------
// fp32 -> fp16 weight conversion
// ---------------------------------------------------------------------------
__global__ void k_cvt_f16(const float* __restrict__ s, _Float16* __restrict__ d, int n) {
  int i = blockIdx.x * blockDim.x + threadIdx.x;
  if (i < n) d[i] = (_Float16)s[i];
}

// ---------------------------------------------------------------------------
// WMMA GEMM:  C[M,N] = act(A[M,K](f32) * W[N,K]^T(f16) + bias)
// Block = 4 waves; wave w owns M-tile (blockIdx.y*4+w)*16; all waves share the
// 64-column strip n_base = blockIdx.x*64.  Per 32-wide K slab the block's
// 64x32 fp16 W panel (4KB) is async-copied to LDS (double buffer) and each
// wave reads its four 16x32 B fragments from LDS (ds_load_b128).
// A fp16 fragment layout (wave32): lane r=lane&15 holds row (mtile+r);
//   hi=lane>>4 selects K-halves; VGPR j holds K pair kb(j,hi), kb+1.
// STORE_MODE: 0 = plain row-major store
//             1 = h[b, (rev?L-1-l:l), n] += v   (out_proj residual, M=b*L+l)
//             2 = h[b, n, d] += v  transposed   (MLP back-scatter, M=b*DM+d)
// ACT: 0 none, 1 relu
// ---------------------------------------------------------------------------
template <int STORE_MODE, int ACT>
__global__ void k_gemm_wmma(const float* __restrict__ A,
                            const _Float16* __restrict__ W,
                            const float* __restrict__ bias,
                            float* __restrict__ C,
                            int M, int N, int K, int rev) {
  __shared__ _Float16 sW[2][64 * 32];   // [buf][row*32 + k] : 4KB per buffer
  const int tid  = threadIdx.x;
  const int lane = tid & 31;
  const int wave = tid >> 5;
  const int r    = lane & 15;
  const int hi   = lane >> 4;
  const int m_tile = (blockIdx.y * 4 + wave) * 16;
  const int n_base = blockIdx.x * 64;

  // per-thread async-copy chunks: 256 x 16B chunks per slab, 2 per thread
  uint64_t gsrc[2];
  uint32_t ldst[2];
  const uint32_t ldsBase = lds_addr_of(&sW[0][0]);
#pragma unroll
  for (int u = 0; u < 2; ++u) {
    const int c = tid + u * 128;
    const int row = c >> 2;          // 0..63  (column n_base+row of C)
    const int sub = c & 3;           // 16B sub-chunk within the 64B slab row
    int nrow = n_base + row;
    if (nrow >= N) nrow = N - 1;     // clamp; dup loads, stores are guarded
    gsrc[u] = (uint64_t)(uintptr_t)(W + (size_t)nrow * K) + (uint64_t)(sub * 16);
    ldst[u] = (uint32_t)(row * 64 + sub * 16);
  }
  auto issue_slab = [&](int k0, int buf) {
#pragma unroll
    for (int u = 0; u < 2; ++u) {
      const uint64_t g = gsrc[u] + (uint64_t)k0 * 2;            // halfs->bytes
      const uint32_t l = ldsBase + (uint32_t)buf * 4096u + ldst[u];
      asm volatile("global_load_async_to_lds_b128 %0, %1, off"
                   :: "v"(l), "v"(g) : "memory");
    }
  };

  v8f acc[4];
#pragma unroll
  for (int t = 0; t < 4; ++t) acc[t] = (v8f){0.f,0.f,0.f,0.f,0.f,0.f,0.f,0.f};

  const float* arow = A + (size_t)(m_tile + r) * K;

  issue_slab(0, 0);
  asm volatile("s_wait_asynccnt 0x0" ::: "memory");
  __syncthreads();

  const int nslab = K >> 5;
  for (int s = 0; s < nslab; ++s) {
    const int k0  = s << 5;
    const int buf = s & 1;
    if (s + 1 < nslab) issue_slab(k0 + 32, buf ^ 1);  // overlap copy & math

    __builtin_prefetch(arow + k0 + 64, 0, 1);         // A slab two ahead
    v16h a;
#pragma unroll
    for (int j = 0; j < 8; ++j) {
      const int kb = ((j & 4) ? (8 + 2 * j) : (2 * j)) + hi * 8;
      float2 f = *(const float2*)(arow + k0 + kb);
      a[2 * j]     = (_Float16)f.x;
      a[2 * j + 1] = (_Float16)f.y;
    }
#pragma unroll
    for (int t = 0; t < 4; ++t) {
      const _Float16* wrow = &sW[buf][(t * 16 + r) * 32];
      v16h b;
#pragma unroll
      for (int j = 0; j < 8; ++j) {
        const int kb = ((j & 4) ? (8 + 2 * j) : (2 * j)) + hi * 8;
        b[2 * j]     = wrow[kb];
        b[2 * j + 1] = wrow[kb + 1];
      }
      acc[t] = __builtin_amdgcn_wmma_f32_16x16x32_f16(
          false, a, false, b, (short)0, acc[t], false, false);
    }
    asm volatile("s_wait_asynccnt 0x0" ::: "memory");
    __syncthreads();
  }

  // C/D layout: col = lane&15, VGPR i -> row m_tile + i + 8*hi
#pragma unroll
  for (int t = 0; t < 4; ++t) {
    const int n = n_base + t * 16 + r;
    if (n >= N) continue;
    const float bv = bias ? bias[n] : 0.f;
#pragma unroll
    for (int i = 0; i < 8; ++i) {
      const int m = m_tile + i + 8 * hi;
      float v = acc[t][i] + bv;
      if (ACT == 1) v = fmaxf(v, 0.f);
      if (STORE_MODE == 0) {
        C[(size_t)m * N + n] = v;
      } else if (STORE_MODE == 1) {
        const int b = m >> 10;
        int       l = m & 1023;
        if (rev) l = 1023 - l;
        float* p = C + ((size_t)((b << 10) | l) * DM + n);
        *p += v;
      } else {
        const int b = m >> 8;
        const int d = m & 255;
        float* p = C + (((size_t)b * L_ + n) * DM + d);
        *p += v;
      }
    }
  }
}

// ---------------------------------------------------------------------------
// LayerNorm over DM=256, with optional time reversal of the source row.
// One block (256 threads = 8 waves) per (b,l) row.
// ---------------------------------------------------------------------------
__global__ void k_ln256(const float* __restrict__ x, const float* __restrict__ w,
                        const float* __restrict__ b, float* __restrict__ y, int rev) {
  __shared__ float s1[8], s2[8];
  const int row = blockIdx.x;
  const int bi = row >> 10, l = row & 1023;
  const int src = (bi << 10) | (rev ? (1023 - l) : l);
  const float v = x[(size_t)src * DM + threadIdx.x];
  const int lane = threadIdx.x & 31, wv = threadIdx.x >> 5;
  float s = wave_sum32(v), q = wave_sum32(v * v);
  if (lane == 0) { s1[wv] = s; s2[wv] = q; }
  __syncthreads();
  float ts = 0.f, tq = 0.f;
#pragma unroll
  for (int i = 0; i < 8; ++i) { ts += s1[i]; tq += s2[i]; }
  const float mean = ts * (1.0f / DM);
  const float var  = tq * (1.0f / DM) - mean * mean;
  y[(size_t)row * DM + threadIdx.x] =
      (v - mean) * rsqrtf(var + 1e-5f) * w[threadIdx.x] + b[threadIdx.x];
}

// ---------------------------------------------------------------------------
// Normalize over L (transposed view) -> tbuf[b,d,l].  LN (layer 0) or RMS.
// One block per (b,d); 4 strided elements per thread.
// ---------------------------------------------------------------------------
__global__ void k_normT(const float* __restrict__ h, const float* __restrict__ nw,
                        const float* __restrict__ nb, float* __restrict__ t,
                        int use_ln) {
  __shared__ float s1[8], s2[8];
  const int bd = blockIdx.x;                // b*DM + d
  const int b = bd >> 8, d = bd & 255;
  const float* base = h + ((size_t)b * L_) * DM + d;
  float vals[4];
  float s = 0.f, q = 0.f;
#pragma unroll
  for (int j = 0; j < 4; ++j) {
    const int l = threadIdx.x + 256 * j;
    const float v = base[(size_t)l * DM];
    vals[j] = v; s += v; q += v * v;
  }
  const int lane = threadIdx.x & 31, wv = threadIdx.x >> 5;
  s = wave_sum32(s); q = wave_sum32(q);
  if (lane == 0) { s1[wv] = s; s2[wv] = q; }
  __syncthreads();
  float ts = 0.f, tq = 0.f;
#pragma unroll
  for (int i = 0; i < 8; ++i) { ts += s1[i]; tq += s2[i]; }
  const float mean = ts * (1.0f / L_);
  float scale, shift_mean;
  if (use_ln) { const float var = tq * (1.0f / L_) - mean * mean;
                scale = rsqrtf(var + 1e-5f); shift_mean = mean; }
  else        { scale = rsqrtf(tq * (1.0f / L_) + 1e-5f); shift_mean = 0.f; }
  float* out = t + (size_t)bd * L_;
#pragma unroll
  for (int j = 0; j < 4; ++j) {
    const int l = threadIdx.x + 256 * j;
    float o = (vals[j] - shift_mean) * scale * nw[l];
    if (use_ln) o += nb[l];
    out[l] = o;
  }
}

// ---------------------------------------------------------------------------
// Depthwise causal conv (K=4) over xa = xz[..., :DI], then SiLU -> xc
// ---------------------------------------------------------------------------
__global__ void k_conv(const float* __restrict__ xz, const float* __restrict__ cw,
                       const float* __restrict__ cb, float* __restrict__ xc) {
  const int idx = blockIdx.x * blockDim.x + threadIdx.x;
  if (idx >= B_ * L_ * DI) return;
  const int d = idx & (DI - 1);
  const int l = (idx >> 9) & (L_ - 1);
  const int b = idx >> 19;
  float acc = cb[d];
#pragma unroll
  for (int k = 0; k < 4; ++k) {
    const int ls = l - 3 + k;
    if (ls >= 0)
      acc += xz[((size_t)((b << 10) | ls)) * (2 * DI) + d] * cw[d * 4 + k];
  }
  xc[idx] = silu_f(acc);
}

// ---------------------------------------------------------------------------
// delta = softplus(xdbl[:, :16] @ dt_w^T + dt_b)     (K=16 -> scalar VALU)
// ---------------------------------------------------------------------------
__global__ void k_dt(const float* __restrict__ xdbl, const float* __restrict__ dtw,
                     const float* __restrict__ dtb, float* __restrict__ delta) {
  const int idx = blockIdx.x * blockDim.x + threadIdx.x;
  if (idx >= B_ * L_ * DI) return;
  const int d = idx & (DI - 1);
  const size_t row = (size_t)(idx >> 9);
  const float* xd = xdbl + row * 48;
  const float* dw = dtw + (size_t)d * DTR;
  float acc = dtb[d];
#pragma unroll
  for (int n = 0; n < DTR; ++n) acc += xd[n] * dw[n];
  delta[idx] = softplus_f(acc);
}

// ---------------------------------------------------------------------------
// Fused selective scan + skip (x*D) + SiLU gate.
// One thread per channel d; 16 h-states in registers; B/C via LDS broadcast.
// Grid: B_*2 blocks of 256 threads (DI=512 channels per batch).
// ---------------------------------------------------------------------------
__global__ void k_scan(const float* __restrict__ delta, const float* __restrict__ xc,
                       const float* __restrict__ xdbl, const float* __restrict__ xz,
                       const float* __restrict__ A_log, const float* __restrict__ Dp,
                       float* __restrict__ yg) {
  __shared__ float sB[DS_], sC[DS_];
  const int b = blockIdx.x >> 1;
  const int d = ((blockIdx.x & 1) << 8) + threadIdx.x;
  float An[DS_];
#pragma unroll
  for (int n = 0; n < DS_; ++n) An[n] = -__expf(A_log[(size_t)d * DS_ + n]);
  const float Dd = Dp[d];
  float hst[DS_];
#pragma unroll
  for (int n = 0; n < DS_; ++n) hst[n] = 0.f;

  for (int l = 0; l < L_; ++l) {
    const size_t row = ((size_t)b << 10) | l;
    if (threadIdx.x < 32) {
      const float v = xdbl[row * 48 + 16 + threadIdx.x];
      if (threadIdx.x < 16) sB[threadIdx.x] = v;
      else                  sC[threadIdx.x - 16] = v;
    }
    __syncthreads();
    const float dt = delta[row * DI + d];
    const float xv = xc[row * DI + d];
    float acc = 0.f;
#pragma unroll
    for (int n = 0; n < DS_; ++n) {
      const float dA = __expf(dt * An[n]);
      hst[n] = dA * hst[n] + (dt * sB[n]) * xv;
      acc += hst[n] * sC[n];
    }
    const float res = xz[row * (2 * DI) + DI + d];
    yg[row * DI + d] = (acc + xv * Dd) * silu_f(res);
    __syncthreads();
  }
}

// ---------------------------------------------------------------------------
__global__ void k_add(float* __restrict__ h, const float* __restrict__ y, int n) {
  const int i = blockIdx.x * blockDim.x + threadIdx.x;
  if (i < n) h[i] += y[i];
}

// ===========================================================================
// Host side
// ===========================================================================
struct MP {
  const float *ln_w, *ln_b, *in_w, *conv_w, *conv_b, *xproj_w,
              *dt_w, *dt_b, *A_log, *Dp, *out_w;
};
struct LP { const float *nw, *w1, *b1, *w2, *b2, *nb; };

static inline size_t a256(size_t x) { return (x + 255) & ~(size_t)255; }

extern "C" void kernel_launch(void* const* d_in, const int* in_sizes, int n_in,
                              void* d_out, int out_size, void* d_ws, size_t ws_size,
                              hipStream_t stream) {
  (void)in_sizes; (void)n_in; (void)out_size; (void)ws_size;
  // ---- input walk: setup_inputs() dict insertion order, params recursively
  //      in their dict insertion order ----
  int p = 0;
  auto nf = [&]() { return (const float*)d_in[p++]; };
  const float* x      = nf();
  const float* emb_w  = nf();
  const float* emb_b  = nf();
  MP M1[NL], M2[NL];
  auto mamba = [&](MP& m) {
    m.ln_w = nf(); m.ln_b = nf(); m.in_w = nf(); m.conv_w = nf(); m.conv_b = nf();
    m.xproj_w = nf(); m.dt_w = nf(); m.dt_b = nf(); m.A_log = nf(); m.Dp = nf();
    m.out_w = nf();
  };
  for (int i = 0; i < NL; ++i) mamba(M1[i]);
  for (int i = 0; i < NL; ++i) mamba(M2[i]);
  LP lin[NL];
  for (int i = 0; i < NL; ++i) {
    lin[i].nw = nf(); lin[i].w1 = nf(); lin[i].b1 = nf();
    lin[i].w2 = nf(); lin[i].b2 = nf();
    lin[i].nb = (i == 0) ? nf() : nullptr;
  }
  const float* normf_w = nf();
  const float* normf_b = nf();
  const float* head_w  = nf();
  const float* head_b  = nf();

  // ---- workspace layout ----
  char* ws = (char*)d_ws;
  size_t off = 0;
  auto allocH = [&](size_t n) { _Float16* r = (_Float16*)(ws + off);
                                off = a256(off + n * sizeof(_Float16)); return r; };
  auto allocF = [&](size_t n) { float* r = (float*)(ws + off);
                                off = a256(off + n * sizeof(float)); return r; };
  _Float16* Whe = allocH((size_t)DM * V_);
  _Float16 *Win1[NL], *Wxp1[NL], *Wout1[NL], *Win2[NL], *Wxp2[NL], *Wout2[NL];
  for (int i = 0; i < NL; ++i) {
    Win1[i] = allocH((size_t)2 * DI * DM); Wxp1[i] = allocH((size_t)48 * DI);
    Wout1[i] = allocH((size_t)DM * DI);
  }
  for (int i = 0; i < NL; ++i) {
    Win2[i] = allocH((size_t)2 * DI * DM); Wxp2[i] = allocH((size_t)48 * DI);
    Wout2[i] = allocH((size_t)DM * DI);
  }
  _Float16 *Ww1[NL], *Ww2[NL];
  for (int i = 0; i < NL; ++i) { Ww1[i] = allocH((size_t)HID_ * L_);
                                 Ww2[i] = allocH((size_t)L_ * HID_); }
  _Float16* Whead = allocH((size_t)V_ * DM);

  float* h    = allocF((size_t)B_ * L_ * DM);
  float* xln  = allocF((size_t)B_ * L_ * DM);
  float* xz   = allocF((size_t)B_ * L_ * 2 * DI);
  float* xc   = allocF((size_t)B_ * L_ * DI);
  float* xdbl = allocF((size_t)B_ * L_ * 48);
  float* delta= allocF((size_t)B_ * L_ * DI);
  float* yg   = allocF((size_t)B_ * L_ * DI);
  float* yo   = allocF((size_t)B_ * L_ * DM);
  float* tbuf = yg;      // free after scans
  float* mhid = delta;   // free after scans
  float* hn   = xln;     // free after last layer

  auto cvt = [&](const float* s, _Float16* d, int n) {
    k_cvt_f16<<<(n + 255) / 256, 256, 0, stream>>>(s, d, n);
  };
  cvt(emb_w, Whe, DM * V_);
  for (int i = 0; i < NL; ++i) {
    cvt(M1[i].in_w, Win1[i], 2 * DI * DM); cvt(M1[i].xproj_w, Wxp1[i], 48 * DI);
    cvt(M1[i].out_w, Wout1[i], DM * DI);
    cvt(M2[i].in_w, Win2[i], 2 * DI * DM); cvt(M2[i].xproj_w, Wxp2[i], 48 * DI);
    cvt(M2[i].out_w, Wout2[i], DM * DI);
    cvt(lin[i].w1, Ww1[i], HID_ * L_); cvt(lin[i].w2, Ww2[i], L_ * HID_);
  }
  cvt(head_w, Whead, V_ * DM);

  auto gemm = [&](const float* A, const _Float16* W, const float* bias, float* C,
                  int M, int N, int K, int mode, int act, int rev) {
    dim3 g((N + 63) / 64, M / 64), blk(128);
    if (mode == 0 && act == 0)
      k_gemm_wmma<0, 0><<<g, blk, 0, stream>>>(A, W, bias, C, M, N, K, rev);
    else if (mode == 0 && act == 1)
      k_gemm_wmma<0, 1><<<g, blk, 0, stream>>>(A, W, bias, C, M, N, K, rev);
    else if (mode == 1)
      k_gemm_wmma<1, 0><<<g, blk, 0, stream>>>(A, W, bias, C, M, N, K, rev);
    else
      k_gemm_wmma<2, 0><<<g, blk, 0, stream>>>(A, W, bias, C, M, N, K, rev);
  };

  const int BL = B_ * L_;                    // 8192
  // embedding: h = x @ emb_w^T + emb_b
  gemm(x, Whe, emb_b, h, BL, DM, V_, 0, 0, 0);

  for (int i = 0; i < NL; ++i) {
    for (int dir = 0; dir < 2; ++dir) {
      const MP& P = dir ? M2[i] : M1[i];
      const _Float16* win  = dir ? Win2[i]  : Win1[i];
      const _Float16* wxp  = dir ? Wxp2[i]  : Wxp1[i];
      const _Float16* wout = dir ? Wout2[i] : Wout1[i];
      k_ln256<<<BL, 256, 0, stream>>>(h, P.ln_w, P.ln_b, xln, dir);
      gemm(xln, win, nullptr, xz, BL, 2 * DI, DM, 0, 0, 0);
      k_conv<<<(BL * DI + 255) / 256, 256, 0, stream>>>(xz, P.conv_w, P.conv_b, xc);
      gemm(xc, wxp, nullptr, xdbl, BL, 48, DI, 0, 0, 0);
      k_dt<<<(BL * DI + 255) / 256, 256, 0, stream>>>(xdbl, P.dt_w, P.dt_b, delta);
      k_scan<<<B_ * 2, 256, 0, stream>>>(delta, xc, xdbl, xz, P.A_log, P.Dp, yg);
      if (dir == 0) gemm(yg, wout, nullptr, yo, BL, DM, DI, 0, 0, 0);
      else          gemm(yg, wout, nullptr, yo, BL, DM, DI, 1, 0, 1);
    }
    k_add<<<(BL * DM + 255) / 256, 256, 0, stream>>>(h, yo, BL * DM);
    k_normT<<<B_ * DM, 256, 0, stream>>>(h, lin[i].nw, lin[i].nb, tbuf, i == 0 ? 1 : 0);
    gemm(tbuf, Ww1[i], lin[i].b1, mhid, B_ * DM, HID_, L_, 0, 1, 0);
    gemm(mhid, Ww2[i], lin[i].b2, h,    B_ * DM, L_, HID_, 2, 0, 0);
  }

  k_ln256<<<BL, 256, 0, stream>>>(h, normf_w, normf_b, hn, 0);
  gemm(hn, Whead, head_b, (float*)d_out, BL, V_, DM, 0, 0, 0);
}
